// TadaConvNeXt_48249662604033
// MI455X (gfx1250) — matrix operand order repaired and checked by reading, not compiled, not run on hardware
//
#include <hip/hip_runtime.h>
#include <hip/hip_bf16.h>
#include <stdint.h>

// ---------------- problem constants ----------------
#define B_   8
#define C_   384
#define T_   8
#define H_   28
#define W_   28
#define HW_  784
#define N_   (B_*T_*HW_)      // 50176
#define CR_  96
#define HID_ 1536
#define MT   64               // rows of N per MLP block
#define FC   64               // hid chunk

typedef __attribute__((ext_vector_type(16))) __bf16 v16bf;
typedef __attribute__((ext_vector_type(8)))  __bf16 bf16x8;
typedef __attribute__((ext_vector_type(8)))  float  v8f;

// ---------------- async global->LDS copy (CDNA5) ----------------
#if defined(__has_builtin)
#if __has_builtin(__builtin_amdgcn_global_load_async_to_lds_b128)
#define HAVE_ASYNC_LDS 1
#endif
#endif

typedef int v4i __attribute__((vector_size(16)));            // matches builtin proto
typedef __attribute__((address_space(1))) v4i v4i_as1;       // global
typedef __attribute__((address_space(3))) v4i v4i_as3;       // LDS

__device__ __forceinline__ void cp16(void* dst_lds, const void* src_glb) {
#ifdef HAVE_ASYNC_LDS
  __builtin_amdgcn_global_load_async_to_lds_b128(
      (v4i_as1*)(uintptr_t)src_glb,
      (v4i_as3*)(unsigned int)(uintptr_t)dst_lds,
      0, 0);
#else
  *(uint4*)dst_lds = *(const uint4*)src_glb;
#endif
}

__device__ __forceinline__ void cp_wait() {
#ifdef HAVE_ASYNC_LDS
#if __has_builtin(__builtin_amdgcn_s_wait_asynccnt)
  __builtin_amdgcn_s_wait_asynccnt(0);
#else
  asm volatile("s_wait_asynccnt 0x0" ::: "memory");
#endif
#endif
}

__device__ __forceinline__ float gelu_exact(float v) {
  return 0.5f * v * (1.0f + erff(v * 0.70710678118654752f));
}

// ---------------- K1: spatial pool  xa[b,c,t] = mean_hw x ----------------
__global__ void k_pool(const float* __restrict__ x, float* __restrict__ xa) {
  int bct = blockIdx.x;                       // (b*C + c)*T + t
  const float* p = x + (size_t)bct * HW_;
  float s = 0.f;
  for (int i = threadIdx.x; i < HW_; i += 128) s += p[i];
  __shared__ float red[128];
  red[threadIdx.x] = s; __syncthreads();
  for (int st = 64; st > 0; st >>= 1) {
    if (threadIdx.x < st) red[threadIdx.x] += red[threadIdx.x + st];
    __syncthreads();
  }
  if (threadIdx.x == 0) xa[bct] = red[0] * (1.0f / HW_);
}

// ---------------- K2a: g linear  glin[b,o] = gb[o] + sum_c mean_t(xa) * gw[o,c] ----------------
__global__ void k_glin(const float* __restrict__ xa, const float* __restrict__ gw,
                       const float* __restrict__ gb, float* __restrict__ glin) {
  int b = blockIdx.x, o = threadIdx.x;        // 384 threads
  __shared__ float gm[C_];
  float s = 0.f;
  for (int t = 0; t < T_; ++t) s += xa[(b * C_ + o) * T_ + t];
  gm[o] = s * (1.0f / T_);
  __syncthreads();
  float acc = gb[o];
  for (int c = 0; c < C_; ++c) acc += gm[c] * gw[o * C_ + c];
  glin[b * C_ + o] = acc;
}

// ---------------- K2b: conv_a + LN + gelu -> hg[b,j,t] ----------------
__global__ void k_routea(const float* __restrict__ xa, const float* __restrict__ glin,
                         const float* __restrict__ aw, const float* __restrict__ ab,
                         const float* __restrict__ lnw, const float* __restrict__ lnb,
                         float* __restrict__ hg) {
  int bt = blockIdx.x; int b = bt / T_, t = bt % T_;
  int j = threadIdx.x;                        // 128 threads, j<96 active
  float s = 0.f;
  if (j < CR_) {
    s = ab[j];
    for (int c = 0; c < C_; ++c) {
      float g = glin[b * C_ + c];
      const float* w = aw + (j * C_ + c) * 3;
      #pragma unroll
      for (int k = 0; k < 3; ++k) {
        int tt = t + k - 1;
        if (tt >= 0 && tt < T_) s += w[k] * (xa[(b * C_ + c) * T_ + tt] + g);
      }
    }
  }
  __shared__ float r1[128], r2[128];
  r1[threadIdx.x] = (j < CR_) ? s : 0.f;
  r2[threadIdx.x] = (j < CR_) ? s * s : 0.f;
  __syncthreads();
  for (int st = 64; st > 0; st >>= 1) {
    if (threadIdx.x < st) { r1[threadIdx.x] += r1[threadIdx.x + st]; r2[threadIdx.x] += r2[threadIdx.x + st]; }
    __syncthreads();
  }
  float u = r1[0] * (1.0f / CR_);
  float var = r2[0] * (1.0f / CR_) - u * u;
  if (j < CR_) {
    float xn = (s - u) * rsqrtf(var + 1e-6f);
    hg[(b * CR_ + j) * T_ + t] = gelu_exact(xn * lnw[j] + lnb[j]);
  }
}

// ---------------- K2c: alpha/beta convs ----------------
__global__ void k_routeb(const float* __restrict__ hg, const float* __restrict__ bw,
                         const float* __restrict__ bbw,
                         float* __restrict__ alpha, float* __restrict__ beta) {
  int bt = blockIdx.x; int b = bt / T_, t = bt % T_;
  int c = threadIdx.x;                        // 384 threads
  float sa = 0.f, sb = 0.f;
  for (int j = 0; j < CR_; ++j) {
    const float* wa = bw  + (c * CR_ + j) * 3;
    const float* wb = bbw + (c * CR_ + j) * 3;
    #pragma unroll
    for (int k = 0; k < 3; ++k) {
      int tt = t + k - 1;
      if (tt >= 0 && tt < T_) {
        float h = hg[(b * CR_ + j) * T_ + tt];
        sa += wa[k] * h; sb += wb[k] * h;
      }
    }
  }
  alpha[(b * C_ + c) * T_ + t] = sa + 1.0f;
  beta [(b * C_ + c) * T_ + t] = sb + 1.0f;
}

// ---------------- K3: convert MLP weights to bf16 ----------------
__global__ void k_tobf16(const float* __restrict__ w1, const float* __restrict__ w2,
                         __bf16* __restrict__ w1b, __bf16* __restrict__ w2b) {
  int i = blockIdx.x * 256 + threadIdx.x;
  if (i < HID_ * C_) { w1b[i] = (__bf16)w1[i]; w2b[i] = (__bf16)w2[i]; }
}

// ---------------- K4a: depthwise 7x7 + alpha/beta -> yv (n, c) f32 ----------------
__global__ void k_dwconv(const float* __restrict__ x, const float* __restrict__ dww,
                         const float* __restrict__ dwb, const float* __restrict__ alpha,
                         const float* __restrict__ beta, float* __restrict__ yv) {
  int bth = blockIdx.x; int h = bth % H_; int bt = bth / H_;
  int b = bt / T_, t = bt % T_;
  for (int it = 0; it < 42; ++it) {           // 42*256 = 384*28
    int idx = it * 256 + threadIdx.x;
    int c = idx / W_, w = idx % W_;
    const float* xp = x + ((size_t)(b * C_ + c) * T_ + t) * HW_;
    const float* wp = dww + c * 49;
    float s = 0.f;
    #pragma unroll
    for (int dy = 0; dy < 7; ++dy) {
      int hh = h + dy - 3; if (hh < 0 || hh >= H_) continue;
      const float* row = xp + hh * W_;
      #pragma unroll
      for (int dx = 0; dx < 7; ++dx) {
        int ww = w + dx - 3; if (ww < 0 || ww >= W_) continue;
        s += row[ww] * wp[dy * 7 + dx];
      }
    }
    float a  = alpha[(b * C_ + c) * T_ + t];
    float be = beta [(b * C_ + c) * T_ + t];
    int n = ((b * T_ + t) * H_ + h) * W_ + w;
    yv[(size_t)n * C_ + c] = a * s + be * dwb[c];
  }
}

// ---------------- K4b: LayerNorm over C -> yn (n, c) bf16 ----------------
__global__ void k_lnorm(const float* __restrict__ yv, const float* __restrict__ nw,
                        const float* __restrict__ nb, __bf16* __restrict__ yn) {
  int n = blockIdx.x;
  const float* p = yv + (size_t)n * C_;
  float v[3], s = 0.f, ss = 0.f;
  #pragma unroll
  for (int k = 0; k < 3; ++k) {
    float t = p[threadIdx.x + k * 128];
    v[k] = t; s += t; ss += t * t;
  }
  __shared__ float r1[128], r2[128];
  r1[threadIdx.x] = s; r2[threadIdx.x] = ss; __syncthreads();
  for (int st = 64; st > 0; st >>= 1) {
    if (threadIdx.x < st) { r1[threadIdx.x] += r1[threadIdx.x + st]; r2[threadIdx.x] += r2[threadIdx.x + st]; }
    __syncthreads();
  }
  float u = r1[0] * (1.0f / C_);
  float var = r2[0] * (1.0f / C_) - u * u;
  float inv = rsqrtf(var + 1e-6f);
  #pragma unroll
  for (int k = 0; k < 3; ++k) {
    int c = threadIdx.x + k * 128;
    yn[(size_t)n * C_ + c] = (__bf16)((v[k] - u) * inv * nw[c] + nb[c]);
  }
}

// ---------------- WMMA fragment loaders (per ISA VGPR layouts) ----------------
// A (16x32 bf16): lane half h, element e<8 -> K = h*8+e ; e>=8 -> K = 16+h*8+(e-8)
__device__ __forceinline__ v16bf ldfragA(const __bf16* p) {
  bf16x8 lo = *(const bf16x8*)p;
  bf16x8 hi = *(const bf16x8*)(p + 16);
  return __builtin_shufflevector(lo, hi, 0,1,2,3,4,5,6,7,8,9,10,11,12,13,14,15);
}
// B (32x16 bf16): lane half h, element e<8 -> K = h*8+e ; e>=8 -> K = 16+h*8+(e-8)
__device__ __forceinline__ v16bf ldfragB(const __bf16* p) {
  bf16x8 lo = *(const bf16x8*)p;
  bf16x8 hi = *(const bf16x8*)(p + 16);
  return __builtin_shufflevector(lo, hi, 0,1,2,3,4,5,6,7,8,9,10,11,12,13,14,15);
}

// ---------------- K5: fused MLP  gelu(yn@W1^T+b1)@W2^T + b2, *gamma -> mo bf16 ----------------
__global__ void __launch_bounds__(256) k_mlp(
    const __bf16* __restrict__ yn, const __bf16* __restrict__ w1b,
    const __bf16* __restrict__ w2b, const float* __restrict__ b1,
    const float* __restrict__ b2, const float* __restrict__ gamma,
    __bf16* __restrict__ mo) {
  __shared__ __align__(16) __bf16 sA [MT * C_];   // 48 KB  yn tile   (m, c)
  __shared__ __align__(16) __bf16 sW1[FC * C_];   // 48 KB  W1 chunk  (f, c)
  __shared__ __align__(16) __bf16 sW2[C_ * FC];   // 48 KB  W2 chunk  (c, f)
  __shared__ __align__(16) __bf16 sH [MT * FC];   //  8 KB  h chunk   (m, f)

  const int tid  = threadIdx.x;
  const int wave = tid >> 5, lane = tid & 31;
  const int half = lane >> 4, l16 = lane & 15;
  const int n0   = blockIdx.x * MT;

  // stage A tile (contiguous 48 KB)
  {
    const __bf16* src = yn + (size_t)n0 * C_;
    for (int i = tid; i < MT * C_ / 8; i += 256) cp16(&sA[i * 8], &src[i * 8]);
  }

  const int mi2 = wave & 3;              // out-tile M row (GEMM2)
  const int njb = (wave >> 2) * 12;      // out-tile N group base
  v8f zero = {};
  v8f acc2[12];
  #pragma unroll
  for (int j = 0; j < 12; ++j) acc2[j] = zero;

  for (int ch = 0; ch < HID_ / FC; ++ch) {
    const int f0 = ch * FC;
    // stage W1 rows (contiguous) and W2 column-chunk
    {
      const __bf16* s1 = w1b + (size_t)f0 * C_;
      for (int i = tid; i < FC * C_ / 8; i += 256) cp16(&sW1[i * 8], &s1[i * 8]);
      for (int u = tid; u < C_ * (FC / 8); u += 256) {
        int c = u / (FC / 8), o = u % (FC / 8);
        cp16(&sW2[c * FC + o * 8], &w2b[(size_t)c * HID_ + f0 + o * 8]);
      }
      cp_wait();
    }
    if (ch + 1 < HID_ / FC)
      __builtin_prefetch(w1b + (size_t)(f0 + FC) * C_, 0, 0);
    __syncthreads();

    // GEMM1: h[64x64] = A[64x384] x W1^T ; 16 tiles, 2 per wave
    #pragma unroll
    for (int q = 0; q < 2; ++q) {
      int tile = wave * 2 + q;
      int hmi = tile >> 2, hfi = tile & 3;
      v8f acc = zero;
      #pragma unroll
      for (int kb = 0; kb < C_ / 32; ++kb) {
        int k0 = kb * 32;
        v16bf a = ldfragA(&sA [(hmi * 16 + l16) * C_ + k0 + half * 8]);
        v16bf b = ldfragB(&sW1[(hfi * 16 + l16) * C_ + k0 + half * 8]);
        acc = __builtin_amdgcn_wmma_f32_16x16x32_bf16(false, a, false, b,
                                                      (short)0, acc, false, false);
      }
      #pragma unroll
      for (int v = 0; v < 8; ++v) {
        int m = hmi * 16 + v + half * 8;
        int f = hfi * 16 + l16;
        float val = acc[v] + b1[f0 + f];
        sH[m * FC + f] = (__bf16)gelu_exact(val);
      }
    }
    __syncthreads();

    // GEMM2 partial: out[64x384] += h[64x64] x W2^T ; 96 tiles, 12 per wave
    #pragma unroll
    for (int j = 0; j < 12; ++j) {
      int ni = njb + j;
      #pragma unroll
      for (int kb = 0; kb < FC / 32; ++kb) {
        int k0 = kb * 32;
        v16bf a = ldfragA(&sH [(mi2 * 16 + l16) * FC + k0 + half * 8]);
        v16bf b = ldfragB(&sW2[(ni  * 16 + l16) * FC + k0 + half * 8]);
        acc2[j] = __builtin_amdgcn_wmma_f32_16x16x32_bf16(false, a, false, b,
                                                          (short)0, acc2[j], false, false);
      }
    }
    __syncthreads();
  }

  // epilogue: (acc + b2) * gamma -> bf16 (n, c)
  #pragma unroll
  for (int j = 0; j < 12; ++j) {
    int c = (njb + j) * 16 + l16;
    #pragma unroll
    for (int v = 0; v < 8; ++v) {
      int m = mi2 * 16 + v + half * 8;
      float val = (acc2[j][v] + b2[c]) * gamma[c];
      mo[(size_t)(n0 + m) * C_ + c] = (__bf16)val;
    }
  }
}

// ---------------- K6: residual, back to (B,C,T,H,W) ----------------
__global__ void k_final(const float* __restrict__ x, const __bf16* __restrict__ mo,
                        float* __restrict__ out) {
  size_t i = (size_t)blockIdx.x * 256 + threadIdx.x;   // < B*C*T*HW
  int thw = (int)(i % (T_ * HW_));
  size_t bc = i / (T_ * HW_);
  int c = (int)(bc % C_);
  int b = (int)(bc / C_);
  int n = b * (T_ * HW_) + thw;
  out[i] = x[i] + (float)mo[(size_t)n * C_ + c];
}

// ---------------- host ----------------
extern "C" void kernel_launch(void* const* d_in, const int* in_sizes, int n_in,
                              void* d_out, int out_size, void* d_ws, size_t ws_size,
                              hipStream_t stream) {
  const float* x     = (const float*)d_in[0];
  const float* dw_w  = (const float*)d_in[1];
  const float* dw_b  = (const float*)d_in[2];
  const float* g_w   = (const float*)d_in[3];
  const float* g_b   = (const float*)d_in[4];
  const float* a_w   = (const float*)d_in[5];
  const float* a_b   = (const float*)d_in[6];
  const float* ln_w  = (const float*)d_in[7];
  const float* ln_b  = (const float*)d_in[8];
  const float* b_w   = (const float*)d_in[9];
  const float* bb_w  = (const float*)d_in[10];
  const float* nw    = (const float*)d_in[11];
  const float* nb    = (const float*)d_in[12];
  const float* pw1w  = (const float*)d_in[13];
  const float* pw1b  = (const float*)d_in[14];
  const float* pw2w  = (const float*)d_in[15];
  const float* pw2b  = (const float*)d_in[16];
  const float* gamma = (const float*)d_in[17];
  float* out = (float*)d_out;

  char* ws = (char*)d_ws;
  size_t off = 0;
  auto take = [&](size_t bytes) -> char* {
    char* p = ws + off;
    off = (off + bytes + 255) & ~(size_t)255;
    return p;
  };
  float*  xa    = (float*)take((size_t)B_ * C_ * T_ * 4);
  float*  glin  = (float*)take((size_t)B_ * C_ * 4);
  float*  hg    = (float*)take((size_t)B_ * CR_ * T_ * 4);
  float*  alpha = (float*)take((size_t)B_ * C_ * T_ * 4);
  float*  beta  = (float*)take((size_t)B_ * C_ * T_ * 4);
  __bf16* w1b   = (__bf16*)take((size_t)HID_ * C_ * 2);
  __bf16* w2b   = (__bf16*)take((size_t)C_ * HID_ * 2);
  __bf16* yn    = (__bf16*)take((size_t)N_ * C_ * 2);
  float*  yv    = (float*)take((size_t)N_ * C_ * 4);
  __bf16* mo    = (__bf16*)yv;   // reuse: yv dead after k_lnorm

  k_pool  <<<B_ * C_ * T_, 128, 0, stream>>>(x, xa);
  k_glin  <<<B_,            C_, 0, stream>>>(xa, g_w, g_b, glin);
  k_routea<<<B_ * T_,      128, 0, stream>>>(xa, glin, a_w, a_b, ln_w, ln_b, hg);
  k_routeb<<<B_ * T_,       C_, 0, stream>>>(hg, b_w, bb_w, alpha, beta);
  k_tobf16<<<(HID_ * C_ + 255) / 256, 256, 0, stream>>>(pw1w, pw2w, w1b, w2b);
  k_dwconv<<<B_ * T_ * H_, 256, 0, stream>>>(x, dw_w, dw_b, alpha, beta, yv);
  k_lnorm <<<N_,           128, 0, stream>>>(yv, nw, nb, yn);
  k_mlp   <<<N_ / MT,      256, 0, stream>>>(yn, w1b, w2b, pw1b, pw2b, gamma, mo);
  k_final <<<(size_t)B_ * C_ * T_ * HW_ / 256, 256, 0, stream>>>(x, mo, out);
  (void)in_sizes; (void)n_in; (void)out_size; (void)ws_size;
}
